// MultiheadAttention_29892972380331
// MI455X (gfx1250) — compile-verified, matrix operand order checked by reading, not run on hardware
//
#include <hip/hip_runtime.h>

// MHA forward, B=2 S=2048 E=1024 H=16 D=64, MI455X (gfx1250).
// f16 WMMA (f32 accum); TDM (tensor_load_to_lds) triple-buffered LDS staging
// with one workgroup barrier per pipeline stage; flash softmax with
// bit-trick OR masking. Buffer rotation is incremental (no % in hot loops).

#define B_ 2
#define S_ 2048
#define E_ 1024
#define H_ 16
#define D_ 64

typedef __attribute__((ext_vector_type(16))) _Float16 v16h;
typedef __attribute__((ext_vector_type(8)))  _Float16 v8h;
typedef __attribute__((ext_vector_type(4)))  _Float16 v4h;
typedef __attribute__((ext_vector_type(8)))  float    v8f;
typedef __attribute__((ext_vector_type(4)))  unsigned v4u;
typedef __attribute__((ext_vector_type(8)))  unsigned v8u;

static __device__ __forceinline__ int rot3(int x) { return x == 2 ? 0 : x + 1; }

// ---- WMMA helpers ---------------------------------------------------------

// Per-lane A/B fragment gather for 16x16x32 f16 WMMA (ISA 7.12.2): lane L is
// row/col L%16; halfs {off..off+7} and {off+16..off+23}, off=(L/16)*8 (caller
// bakes "+(L>>4)*8" into `base`).
static __device__ __forceinline__ v16h load_frag(const _Float16* base) {
  v8h lo = *(const v8h*)(base);
  v8h hi = *(const v8h*)(base + 16);
  v16h r;
#pragma unroll
  for (int i = 0; i < 8; ++i) { r[i] = lo[i]; r[i + 8] = hi[i]; }
  return r;
}

static __device__ __forceinline__ v8f wmma16(v16h a, v16h b, v8f c) {
  return __builtin_amdgcn_wmma_f32_16x16x32_f16(
      false, a, false, b, (short)0, c, false, false);
}

static __device__ __forceinline__ v8h cvt8(v8f x) {
  v8h r;
#pragma unroll
  for (int i = 0; i < 8; ++i) r[i] = (_Float16)x[i];
  return r;
}

// ---- Tensor Data Mover ----------------------------------------------------
// 2D TENSOR_LOAD_TO_LDS: D# group0 (4 SGPRs) + group1 (8 SGPRs) per CDNA5 ISA
// §8.3/§8.4, 2-group form (VADDR2/3=NULL, tensors up to 2D). LDS padding
// (pad every 2^(padIntCode+1) dwords by (padAmtCode+1) dwords) gives a
// bank-conflict-free pitch. tile_d0/stride0 in f16 elements.
static __device__ __forceinline__ void tdm_load_2d(
    unsigned lds_byte_off, const void* gaddr,
    unsigned tile_d0, unsigned tile_d1, unsigned long long stride0,
    unsigned padIntCode, unsigned padAmtCode) {
  unsigned long long ga = (unsigned long long)(size_t)gaddr;
  v4u g0;
  g0[0] = 1u;                                   // count=1 (valid user D#)
  g0[1] = lds_byte_off;                         // lds_addr
  g0[2] = (unsigned)ga;                         // global_addr[31:0]
  g0[3] = (unsigned)(ga >> 32) | 0x80000000u;   // addr[56:32] | type=2
  v8u g1;
  g1[0] = (1u << 16)                            // data_size = 2 bytes
        | (1u << 20)                            // pad_enable
        | (padIntCode << 22) | (padAmtCode << 25);
  g1[1] = (tile_d0 & 0xFFFFu) << 16;            // tensor_dim0[15:0] @ bit48
  g1[2] = (tile_d0 >> 16) | ((tile_d1 & 0xFFFFu) << 16);
  g1[3] = (tile_d1 >> 16) | (tile_d0 << 16);    // dim1 hi | tile_dim0
  g1[4] = tile_d1 & 0xFFFFu;                    // tile_dim1 | tile_dim2=0
  g1[5] = (unsigned)stride0;                    // tensor_dim0_stride lo
  g1[6] = (unsigned)(stride0 >> 32) & 0xFFFFu;
  g1[7] = 0u;
  asm volatile("tensor_load_to_lds %0, %1" :: "s"(g0), "s"(g1) : "memory");
}

static __device__ __forceinline__ unsigned lds_off(const void* p) {
  return (unsigned)(size_t)p;  // flat LDS ptr: low 32 bits = LDS byte offset
}

// ---- format conversion ----------------------------------------------------
__global__ __launch_bounds__(256) void cvt_f32_f16(
    const float* __restrict__ in, _Float16* __restrict__ out, int n) {
  int i = (blockIdx.x * 256 + threadIdx.x) * 4;
  if (i >= n) return;
  float4 v = *(const float4*)(in + i);
  v4h o = {(_Float16)v.x, (_Float16)v.y, (_Float16)v.z, (_Float16)v.w};
  *(v4h*)(out + i) = o;
}

// mask -> additive score bias (exp2 domain). Values are exactly {0, NEG2};
// "OR of padded positions" == bitwise OR of the f32 encodings (see attn).
__global__ __launch_bounds__(256) void mask_to_add(
    const int* __restrict__ m, float* __restrict__ ma, int n) {
  int i = blockIdx.x * 256 + threadIdx.x;
  if (i < n) ma[i] = (m[i] == 0) ? (-1.0e10f * 1.44269504089f) : 0.0f;
}

// ---------------------------------------------------------------------------
// C[M,N] = A[M,K] * B[N,K]^T. Triple-buffered TDM panels, one barrier/step.
// Block 256 thr = 8 waves (4Mx2N), wave tile 32x64 (2x4 WMMA), block 128x128.
// LDS pitch 40 halfs (32 data + 8 pad): rows map to distinct bank groups.
// MODE 0: fp32 out[M,N] + bias[n]
// MODE 1: A=weights(m=ch), B=X(n=token) -> f16 [B,H,S,D]   (Q,K)
// MODE 2: A=X(m=token), B=weights(n=ch) -> f16 [B,H,D,S]   (V^T)
// ---------------------------------------------------------------------------
#define GP 40

template <int MODE>
__global__ __launch_bounds__(256) void gemm_wmma_kernel(
    const _Float16* __restrict__ A, const _Float16* __restrict__ Bm,
    const float* __restrict__ bias, void* __restrict__ outp,
    int M, int N, int K) {
  __shared__ _Float16 sA[3][128 * GP];
  __shared__ _Float16 sB[3][128 * GP];

  const int lane = threadIdx.x & 31;
  const int wave = threadIdx.x >> 5;
  const int lr   = lane & 15;
  const int lh8  = (lane >> 4) << 3;
  const int mBlk = blockIdx.x * 128;
  const int nBlk = blockIdx.y * 128;
  const int mW   = (wave >> 1) * 32;
  const int nW   = (wave & 1) * 64;
  const int kSteps = K >> 5;

  auto issue = [&](int bufIdx, int s) {
    const int kk = s << 5;
    tdm_load_2d(lds_off(sA[bufIdx]), A + (size_t)mBlk * K + kk,
                32, 128, (unsigned)K, 3, 3);
    tdm_load_2d(lds_off(sB[bufIdx]), Bm + (size_t)nBlk * K + kk,
                32, 128, (unsigned)K, 3, 3);
  };

  if (wave == 0) { issue(0, 0); if (kSteps > 1) issue(1, 1); }

  v8f acc[2][4] = {};
  int buf = 0, pbuf = 2;  // compute buffer, prefetch-target buffer

  for (int s = 0; s < kSteps; ++s) {
    if (wave == 0) {
      if (s + 1 < kSteps) __builtin_amdgcn_s_wait_tensorcnt(2);  // pair s done
      else                __builtin_amdgcn_s_wait_tensorcnt(0);
    }
    __syncthreads();
    // stage s+2 target buffer was last read at step s-1 (pre-barrier): safe.
    if (wave == 0 && s + 2 < kSteps) issue(pbuf, s + 2);

    v16h fa[2], fb[4];
#pragma unroll
    for (int mt = 0; mt < 2; ++mt)
      fa[mt] = load_frag(&sA[buf][(mW + mt * 16 + lr) * GP + lh8]);
#pragma unroll
    for (int nt = 0; nt < 4; ++nt)
      fb[nt] = load_frag(&sB[buf][(nW + nt * 16 + lr) * GP + lh8]);
#pragma unroll
    for (int mt = 0; mt < 2; ++mt)
#pragma unroll
      for (int nt = 0; nt < 4; ++nt)
        acc[mt][nt] = wmma16(fa[mt], fb[nt], acc[mt][nt]);

    buf = rot3(buf);
    pbuf = rot3(pbuf);
  }

#pragma unroll
  for (int mt = 0; mt < 2; ++mt) {
#pragma unroll
    for (int nt = 0; nt < 4; ++nt) {
      const int rowb = mBlk + mW + mt * 16 + lh8;  // 8 consecutive M rows
      const int col  = nBlk + nW + nt * 16 + lr;   // fixed N column
      v8f c = acc[mt][nt];
      if (MODE == 0) {
        float* out = (float*)outp;
        const float bv = bias[col];
#pragma unroll
        for (int i = 0; i < 8; ++i)
          out[(size_t)(rowb + i) * N + col] = c[i] + bv;
      } else if (MODE == 1) {
        _Float16* out = (_Float16*)outp;
        const int b = col >> 11, sIdx = col & (S_ - 1);
        const int h = rowb >> 6, d0 = rowb & 63;
#pragma unroll
        for (int i = 0; i < 8; ++i) c[i] += bias[rowb + i];
        *(v8h*)(out + ((size_t)((b * H_ + h) * S_ + sIdx) << 6) + d0) = cvt8(c);
      } else {
        _Float16* out = (_Float16*)outp;
        const int b = rowb >> 11, s0 = rowb & (S_ - 1);
        const int h = col >> 6, d = col & 63;
        c = c + bias[col];
        *(v8h*)(out + ((size_t)((b * H_ + h) * D_ + d) << 11) + s0) = cvt8(c);
      }
    }
  }
}

// ---------------------------------------------------------------------------
// Flash attention. One block = one (b,h) x 128 queries (8 waves x 16-q tile).
// Triple-buffered TDM staging of K (32x64, pitch 72) and V^T (64x32, pitch
// 40) chunks; one barrier per 32-key chunk. Scores computed transposed
// (S^T = K.Q^T): softmax is per-lane + one shfl_xor(16); S^T C-frag key order
// equals the PV B-frag layout -> P repacked to f16 in-lane. O accumulated
// transposed (O^T = V^T @ P^T), contiguous f16 stores to [B,S,E].
// ---------------------------------------------------------------------------
#define KP 72
#define VP 40

__global__ __launch_bounds__(256) void attn_kernel(
    const _Float16* __restrict__ Q, const _Float16* __restrict__ Kf,
    const _Float16* __restrict__ Vt, const float* __restrict__ maskadd,
    _Float16* __restrict__ attn) {
  __shared__ _Float16 sK[3][32 * KP];
  __shared__ _Float16 sV[3][64 * VP];

  const int lane = threadIdx.x & 31;
  const int wave = threadIdx.x >> 5;
  const int bh   = blockIdx.x >> 4;
  const int qblk = blockIdx.x & 15;
  const int b    = bh >> 4;
  const int h    = bh & 15;
  const int lr   = lane & 15;
  const int lh8  = (lane >> 4) << 3;

  const _Float16* Qb = Q  + (size_t)bh * S_ * D_;
  const _Float16* Kb = Kf + (size_t)bh * S_ * D_;
  const _Float16* Vb = Vt + (size_t)bh * D_ * S_;
  const float*    mb = maskadd + b * S_;

  const int q = qblk * 128 + wave * 16 + lr;
  const _Float16* qrow = Qb + (size_t)q * D_;
  const v16h qf0 = load_frag(qrow + 0  + lh8);
  const v16h qf1 = load_frag(qrow + 32 + lh8);
  const unsigned oq = __float_as_uint(mb[q]);  // {0, NEG2} bit pattern

  auto issue = [&](int bufIdx, int c) {
    const int kk = c << 5;
    tdm_load_2d(lds_off(sK[bufIdx]), Kb + (size_t)kk * D_, 64, 32, 64u, 4, 3);
    tdm_load_2d(lds_off(sV[bufIdx]), Vb + kk, 32, 64, (unsigned)S_, 3, 3);
  };
  if (wave == 0) { issue(0, 0); issue(1, 1); }

  v8f oacc[4] = {};
  float mrun = -INFINITY, lrun = 0.0f;
  const float sc = 0.125f * 1.44269504089f;  // 1/sqrt(D) * log2(e)
  const int nCh = S_ >> 5;                   // 64 chunks of 32 keys
  int buf = 0, pbuf = 2;

  for (int c = 0; c < nCh; ++c) {
    const int k0 = c << 5;
    if (wave == 0) {
      if (c + 1 < nCh) __builtin_amdgcn_s_wait_tensorcnt(2);
      else             __builtin_amdgcn_s_wait_tensorcnt(0);
    }
    __syncthreads();
    if (wave == 0 && c + 2 < nCh) issue(pbuf, c + 2);

    // scores^T: two 16-key tiles x (D=64 -> 2 WMMAs each)
    const _Float16* kr0 = &sK[buf][lr * KP + lh8];
    const _Float16* kr1 = &sK[buf][(16 + lr) * KP + lh8];
    v8f s0 = {}, s1 = {};
    s0 = wmma16(load_frag(kr0),      qf0, s0);
    s0 = wmma16(load_frag(kr0 + 32), qf1, s0);
    s1 = wmma16(load_frag(kr1),      qf0, s1);
    s1 = wmma16(load_frag(kr1 + 32), qf1, s1);

    // OR-mask: fmin over {0,NEG2} values == bitwise OR of f32 encodings.
    const float* mk0 = mb + k0 + lh8;
    const float* mk1 = mk0 + 16;
    float t0[8], t1[8];
    float cmax = -INFINITY;
#pragma unroll
    for (int i = 0; i < 8; ++i) {
      t0[i] = s0[i] * sc + __uint_as_float(oq | __float_as_uint(mk0[i]));
      t1[i] = s1[i] * sc + __uint_as_float(oq | __float_as_uint(mk1[i]));
      cmax = fmaxf(cmax, fmaxf(t0[i], t1[i]));
    }
    cmax = fmaxf(cmax, __shfl_xor(cmax, 16, 32));  // partner lane, same q
    const float mnew  = fmaxf(mrun, cmax);
    const float alpha = exp2f(mrun - mnew);
    float psum = 0.0f;
    v16h pf;
#pragma unroll
    for (int i = 0; i < 8; ++i) {
      float p0 = exp2f(t0[i] - mnew);
      float p1 = exp2f(t1[i] - mnew);
      psum += p0 + p1;
      pf[i]     = (_Float16)p0;
      pf[i + 8] = (_Float16)p1;
    }
    psum += __shfl_xor(psum, 16, 32);
    lrun = lrun * alpha + psum;
    mrun = mnew;

#pragma unroll
    for (int t = 0; t < 4; ++t) oacc[t] = oacc[t] * alpha;
#pragma unroll
    for (int t = 0; t < 4; ++t) {
      const _Float16* vr = &sV[buf][(t * 16 + lr) * VP + lh8];
      oacc[t] = wmma16(load_frag(vr), pf, oacc[t]);
    }

    buf = rot3(buf);
    pbuf = rot3(pbuf);
  }

  const float inv = 1.0f / lrun;
#pragma unroll
  for (int t = 0; t < 4; ++t) {
    v8h o = cvt8(oacc[t] * inv);
    _Float16* orow =
        attn + ((size_t)(b * S_ + q) << 10) + h * 64 + t * 16 + lh8;
    *(v8h*)orow = o;
  }
}

// ---------------------------------------------------------------------------
extern "C" void kernel_launch(void* const* d_in, const int* in_sizes, int n_in,
                              void* d_out, int out_size, void* d_ws,
                              size_t ws_size, hipStream_t stream) {
  (void)in_sizes; (void)n_in; (void)out_size; (void)ws_size;
  const float* q   = (const float*)d_in[0];
  const float* ky  = (const float*)d_in[1];
  const float* vl  = (const float*)d_in[2];
  const int*   msk = (const int*)d_in[3];
  const float* Wq  = (const float*)d_in[4];
  const float* bq  = (const float*)d_in[5];
  const float* Wk  = (const float*)d_in[6];
  const float* bk  = (const float*)d_in[7];
  const float* Wv  = (const float*)d_in[8];
  const float* bv  = (const float*)d_in[9];
  const float* Wo  = (const float*)d_in[10];
  const float* bo  = (const float*)d_in[11];
  float* out = (float*)d_out;

  const size_t NX = (size_t)B_ * S_ * E_;
  const size_t NW = (size_t)E_ * E_;

  char* w = (char*)d_ws;
  _Float16* hXq = (_Float16*)w; w += NX * 2;
  _Float16* hXk = (_Float16*)w; w += NX * 2;
  _Float16* hXv = (_Float16*)w; w += NX * 2;
  _Float16* hWq = (_Float16*)w; w += NW * 2;
  _Float16* hWk = (_Float16*)w; w += NW * 2;
  _Float16* hWv = (_Float16*)w; w += NW * 2;
  _Float16* hWo = (_Float16*)w; w += NW * 2;
  _Float16* fQ  = (_Float16*)w; w += NX * 2;  // [B,H,S,D]
  _Float16* fK  = (_Float16*)w; w += NX * 2;  // [B,H,S,D]
  _Float16* fVt = (_Float16*)w; w += NX * 2;  // [B,H,D,S]
  _Float16* fAt = (_Float16*)w; w += NX * 2;  // [B,S,E]
  float*    madd = (float*)w;                  // [B,S]

  dim3 blk(256);
  cvt_f32_f16<<<(int)(NX / 1024), blk, 0, stream>>>(q,  hXq, (int)NX);
  cvt_f32_f16<<<(int)(NX / 1024), blk, 0, stream>>>(ky, hXk, (int)NX);
  cvt_f32_f16<<<(int)(NX / 1024), blk, 0, stream>>>(vl, hXv, (int)NX);
  cvt_f32_f16<<<(int)(NW / 1024), blk, 0, stream>>>(Wq, hWq, (int)NW);
  cvt_f32_f16<<<(int)(NW / 1024), blk, 0, stream>>>(Wk, hWk, (int)NW);
  cvt_f32_f16<<<(int)(NW / 1024), blk, 0, stream>>>(Wv, hWv, (int)NW);
  cvt_f32_f16<<<(int)(NW / 1024), blk, 0, stream>>>(Wo, hWo, (int)NW);
  mask_to_add<<<(B_ * S_) / 256, blk, 0, stream>>>(msk, madd, B_ * S_);

  gemm_wmma_kernel<1><<<dim3(E_ / 128, (B_ * S_) / 128), blk, 0, stream>>>(
      hWq, hXq, bq, fQ, E_, B_ * S_, E_);
  gemm_wmma_kernel<1><<<dim3(E_ / 128, (B_ * S_) / 128), blk, 0, stream>>>(
      hWk, hXk, bk, fK, E_, B_ * S_, E_);
  gemm_wmma_kernel<2><<<dim3((B_ * S_) / 128, E_ / 128), blk, 0, stream>>>(
      hXv, hWv, bv, fVt, B_ * S_, E_, E_);

  attn_kernel<<<B_ * H_ * (S_ / 128), blk, 0, stream>>>(
      fQ, fK, fVt, madd, fAt);

  gemm_wmma_kernel<0><<<dim3((B_ * S_) / 128, E_ / 128), blk, 0, stream>>>(
      fAt, hWo, bo, out, B_ * S_, E_, E_);
}
